// ParaSymbolsTokenModel1_15307263443323
// MI455X (gfx1250) — matrix operand order repaired and checked by reading, not compiled, not run on hardware
//
#include <hip/hip_runtime.h>
#include <math.h>
#include <stdint.h>

// ---------------- model constants ----------------
constexpr int kD   = 4096;
constexpr int kDKV = 1024;
constexpr int kDFF = 14336;
constexpr int kNH  = 32;
constexpr int kNKV = 8;
constexpr int kHD  = 128;
constexpr int kS   = 512;
constexpr int kB   = 4;
constexpr int kTok = kB * kS;          // 2048

typedef __attribute__((ext_vector_type(16))) _Float16 v16h;
typedef __attribute__((ext_vector_type(8)))  _Float16 v8h;
typedef __attribute__((ext_vector_type(8)))  float    v8f;

union Frag { v16h v; v8h h[2]; };

__constant__ float NF4_CODE[16] = {
    -1.0f, -0.6961928009986877f, -0.5250730514526367f, -0.39491748809814453f,
    -0.28444138169288635f, -0.18477343022823334f, -0.09105003625154495f, 0.0f,
    0.07958029955625534f, 0.16093020141124725f, 0.24611230194568634f,
    0.33791524171829224f, 0.44070982933044434f, 0.5626170039176941f,
    0.7229568362236023f, 1.0f};

// ---------------- small helpers ----------------
__device__ inline float waveReduceSum(float v) {
#pragma unroll
  for (int m = 16; m > 0; m >>= 1) v += __shfl_xor(v, m, 32);
  return v;
}
__device__ inline int waveReduceSumI(int v) {
#pragma unroll
  for (int m = 16; m > 0; m >>= 1) v += __shfl_xor(v, m, 32);
  return v;
}

// CDNA5 async global->LDS copy (tracked by ASYNCcnt, not LOADcnt).
// the LDS address comes from the VDST VGPR; generic LDS pointer's low 32 bits are the LDS
// offset, so truncation yields the VDST operand.
__device__ inline void async_copy_b128(const void* gsrc, void* ldst) {
  unsigned int loff = (unsigned int)(uintptr_t)ldst;
  asm volatile("global_load_async_to_lds_b128 %0, %1, off"
               :
               : "v"(loff), "v"((unsigned long long)(uintptr_t)gsrc)
               : "memory");
}
__device__ inline void wait_async0() {
  asm volatile("s_wait_asynccnt 0x0" ::: "memory");
}

// ---------------- embed ----------------
__global__ __launch_bounds__(256) void embed_kernel(
    const int* __restrict__ ids, const float* __restrict__ table,
    float* __restrict__ h) {
  int tok = blockIdx.x;
  const float* row = table + (size_t)ids[tok] * kD;
  float* out = h + (size_t)tok * kD;
  for (int i = threadIdx.x; i < kD; i += 256) out[i] = row[i];
}

// ---------------- rmsnorm (f32 in -> f16 out) ----------------
__global__ __launch_bounds__(256) void rmsnorm_kernel(
    const float* __restrict__ h, const float* __restrict__ w,
    _Float16* __restrict__ out) {
  __shared__ float red[8];
  int tid = threadIdx.x, lane = tid & 31, wv = tid >> 5;
  const float* x = h + (size_t)blockIdx.x * kD;
  float ss = 0.f;
  for (int i = tid; i < kD; i += 256) { float v = x[i]; ss += v * v; }
  ss = waveReduceSum(ss);
  if (lane == 0) red[wv] = ss;
  __syncthreads();
  if (tid == 0) {
    float t = 0.f;
#pragma unroll
    for (int i = 0; i < 8; ++i) t += red[i];
    red[0] = rsqrtf(t / (float)kD + 1e-5f);
  }
  __syncthreads();
  float scale = red[0];
  _Float16* o = out + (size_t)blockIdx.x * kD;
  for (int i = tid; i < kD; i += 256) o[i] = (_Float16)(x[i] * scale * w[i]);
}

// ---------------- NF4 dequant GEMM: out[M,N] = act[M,K] @ W[N,K]^T ----------
// Double-buffered LDS; A tiles staged with async global->LDS (ASYNCcnt);
// NF4 idx stream register-pipelined; NF4 code table wave-resident, looked up
// with ds_bpermute (__shfl) so the hot loop has zero table memory traffic.
// mode 0: store f16   mode 1: accumulate into f32 residual
constexpr int BM = 128, BN = 128, BK = 32;

__global__ __launch_bounds__(256) void gemm_nf4_kernel(
    const _Float16* __restrict__ A, const int* __restrict__ idx,
    const float* __restrict__ absmax, void* __restrict__ outp,
    int N, int K, int mode) {
  __shared__ _Float16 lA[2][BM][BK];   // 2 x 8 KB activations
  __shared__ _Float16 lB[2][BN][BK];   // 2 x 8 KB dequantized weights
  int tid = threadIdx.x, lane = tid & 31, wv = tid >> 5;
  int m0 = blockIdx.y * BM, n0 = blockIdx.x * BN;
  int wm = wv & 3;   // 4 row groups of 32
  int wn = wv >> 2;  // 2 col groups of 64
  int g = lane >> 4, ln = lane & 15;

  // wave-resident NF4 table: lane i holds code[i & 15]; one constant load.
  float codeLane = NF4_CODE[lane & 15];

  // staging coordinates
  int ar = tid >> 2, ac = (tid & 3) * 8;     // A: 2 b128 chunks per thread
  int nl = tid >> 1, kh = (tid & 1) << 4;    // B: one 16-wide k run per thread

  int4 ireg[4];
  float am;

  v8f zero = {0.f, 0.f, 0.f, 0.f, 0.f, 0.f, 0.f, 0.f};
  v8f acc[2][4];
#pragma unroll
  for (int mt = 0; mt < 2; ++mt)
#pragma unroll
    for (int nt = 0; nt < 4; ++nt) acc[mt][nt] = zero;

  // ---- prologue: stage tile 0 ----
  {
    const _Float16* s0 = A + (size_t)(m0 + ar) * K + ac;
    async_copy_b128(s0, &lA[0][ar][ac]);
    async_copy_b128(s0 + (size_t)64 * K, &lA[0][ar + 64][ac]);
    size_t base = (size_t)(n0 + nl) * K + kh;
    const int4* ip = (const int4*)(idx + base);
#pragma unroll
    for (int j = 0; j < 4; ++j) ireg[j] = ip[j];
    am = absmax[base >> 6];
    __builtin_prefetch(idx + base + BK, 0, 3);
  }

  int nk = K / BK;
  for (int t = 0; t < nk; ++t) {
    int cur = t & 1, nxt = cur ^ 1;
    int k0 = t * BK;

    // dequantize current B run: nibble -> ds_bpermute table gather -> scale
    {
      union { v8h h[2]; _Float16 e[16]; } tq;
#pragma unroll
      for (int j = 0; j < 4; ++j) {
        tq.e[4 * j + 0] = (_Float16)(__shfl(codeLane, ireg[j].x & 15, 32) * am);
        tq.e[4 * j + 1] = (_Float16)(__shfl(codeLane, ireg[j].y & 15, 32) * am);
        tq.e[4 * j + 2] = (_Float16)(__shfl(codeLane, ireg[j].z & 15, 32) * am);
        tq.e[4 * j + 3] = (_Float16)(__shfl(codeLane, ireg[j].w & 15, 32) * am);
      }
      *(v8h*)&lB[cur][nl][kh]     = tq.h[0];
      *(v8h*)&lB[cur][nl][kh + 8] = tq.h[1];
    }
    wait_async0();        // A[cur] resident in LDS
    __syncthreads();      // all waves' lB[cur] stored + lA[cur] complete

    // kick off next tile: async A copy + idx register loads
    if (t + 1 < nk) {
      int k1 = k0 + BK;
      const _Float16* s0 = A + (size_t)(m0 + ar) * K + k1 + ac;
      async_copy_b128(s0, &lA[nxt][ar][ac]);
      async_copy_b128(s0 + (size_t)64 * K, &lA[nxt][ar + 64][ac]);
      size_t base = (size_t)(n0 + nl) * K + k1 + kh;
      const int4* ip = (const int4*)(idx + base);
#pragma unroll
      for (int j = 0; j < 4; ++j) ireg[j] = ip[j];
      am = absmax[base >> 6];
      __builtin_prefetch(idx + base + BK, 0, 3);
    }

    // fragments + 8 wmma per wave on buffer `cur`
    Frag a[2];
#pragma unroll
    for (int mt = 0; mt < 2; ++mt) {
      int m = wm * 32 + mt * 16 + ln;
      a[mt].h[0] = *(const v8h*)&lA[cur][m][g * 8];       // K = 8g .. 8g+7
      a[mt].h[1] = *(const v8h*)&lA[cur][m][16 + g * 8];  // K = 16+8g .. +7
    }
#pragma unroll
    for (int nt = 0; nt < 4; ++nt) {
      Frag bfr;
      int n = wn * 64 + nt * 16 + ln;
      bfr.h[0] = *(const v8h*)&lB[cur][n][g * 8];         // K = 8g .. 8g+7
      bfr.h[1] = *(const v8h*)&lB[cur][n][16 + g * 8];    // K = 16+8g .. +7
#pragma unroll
      for (int mt = 0; mt < 2; ++mt)
        acc[mt][nt] = __builtin_amdgcn_wmma_f32_16x16x32_f16(
            false, a[mt].v, false, bfr.v, (short)0, acc[mt][nt], false, false);
    }
    // no trailing barrier needed: buffer `cur` is next rewritten by the lA copies of
    // iteration t+1 and the lB stores of t+2, both ordered after barrier(t+1), which is ordered
    // after every wave's compute(t).
  }

  // epilogue (C layout: lane col = ln, rows 8g+i)
#pragma unroll
  for (int mt = 0; mt < 2; ++mt)
#pragma unroll
    for (int nt = 0; nt < 4; ++nt)
#pragma unroll
      for (int i = 0; i < 8; ++i) {
        int m = m0 + wm * 32 + mt * 16 + g * 8 + i;
        int n = n0 + wn * 64 + nt * 16 + ln;
        if (mode == 0)
          ((_Float16*)outp)[(size_t)m * N + n] = (_Float16)acc[mt][nt][i];
        else
          ((float*)outp)[(size_t)m * N + n] += acc[mt][nt][i];
      }
}

// ---------------- RoPE (in place, f16) ----------------
__global__ void rope_kernel(_Float16* __restrict__ x, int nheads) {
  int tok = blockIdx.x, hh = blockIdx.y;
  int s = tok & (kS - 1);
  _Float16* p = x + ((size_t)tok * nheads + hh) * kHD;
  int i = threadIdx.x;  // 64 threads = pairs
  float inv = __expf(-(float)i * (9.210340371976184f / 64.f));  // 10000^(-i/64)
  float f = (float)s * inv;
  float c = __cosf(f), sn = __sinf(f);
  float x1 = (float)p[i], x2 = (float)p[i + 64];
  p[i]      = (_Float16)(x1 * c - x2 * sn);
  p[i + 64] = (_Float16)(x2 * c + x1 * sn);
}

// ---------------- attention: causal GQA, WMMA for QK^T and PV ------------
__global__ __launch_bounds__(256) void attn_kernel(
    const _Float16* __restrict__ qb, const _Float16* __restrict__ kb,
    const _Float16* __restrict__ vb, const int* __restrict__ amask,
    _Float16* __restrict__ ao) {
  __shared__ float    sc[16][kS];       // 32 KB scores
  __shared__ _Float16 pb[16][kS];       // 16 KB probabilities
  int tid = threadIdx.x, lane = tid & 31, wv = tid >> 5;
  int q0 = blockIdx.x * 16, hh = blockIdx.y, b = blockIdx.z;
  int kvh = hh >> 2;                    // NH/NKV = 4
  int g = lane >> 4, ln = lane & 15;

  // Q fragments: rows q0..q0+15, full HD=128 split in 4 chunks of 32
  Frag qf[4];
  const _Float16* qrow = qb + ((size_t)(b * kS + q0 + ln)) * kD + hh * kHD;
#pragma unroll
  for (int f = 0; f < 4; ++f) {
    qf[f].h[0] = *(const v8h*)(qrow + f * 32 + g * 8);
    qf[f].h[1] = *(const v8h*)(qrow + f * 32 + 16 + g * 8);
  }
  const float scale = 0.08838834764831845f;  // 1/sqrt(128)

  // Pass A: wave wv covers keys [wv*64, wv*64+64)
  for (int kt = 0; kt < 4; ++kt) {
    int kbase = wv * 64 + kt * 16;
    const _Float16* krow =
        kb + ((size_t)(b * kS + kbase + ln) * kNKV + kvh) * kHD;
    v8f acc = {0.f, 0.f, 0.f, 0.f, 0.f, 0.f, 0.f, 0.f};
#pragma unroll
    for (int f = 0; f < 4; ++f) {
      Frag kf;
      kf.h[0] = *(const v8h*)(krow + f * 32 + g * 8);
      kf.h[1] = *(const v8h*)(krow + f * 32 + 16 + g * 8);
      acc = __builtin_amdgcn_wmma_f32_16x16x32_f16(
          false, qf[f].v, false, kf.v, (short)0, acc, false, false);
    }
    int kidx = kbase + ln;
    int mvalid = amask[b * kS + kidx] > 0;
#pragma unroll
    for (int i = 0; i < 8; ++i) {
      int qi = q0 + g * 8 + i;
      sc[g * 8 + i][kidx] =
          (mvalid && kidx <= qi) ? acc[i] * scale : -1e9f;
    }
  }
  __syncthreads();

  // softmax: 16-thread team per query row (contiguous lanes -> shfl works)
  {
    int r = tid >> 4, c = tid & 15;
    float mx = -1e30f;
    for (int j = c; j < kS; j += 16) mx = fmaxf(mx, sc[r][j]);
#pragma unroll
    for (int m = 8; m > 0; m >>= 1) mx = fmaxf(mx, __shfl_xor(mx, m, 32));
    float sum = 0.f;
    for (int j = c; j < kS; j += 16) sum += __expf(sc[r][j] - mx);
#pragma unroll
    for (int m = 8; m > 0; m >>= 1) sum += __shfl_xor(sum, m, 32);
    float inv = 1.0f / sum;
    for (int j = c; j < kS; j += 16)
      pb[r][j] = (_Float16)(__expf(sc[r][j] - mx) * inv);
  }
  __syncthreads();

  // Pass B: O = P @ V ; wave wv owns hd columns [wv*16, wv*16+16)
  int hd = wv * 16 + ln;
  int klim = (q0 + 16 + 31) & ~31;
  if (klim > kS) klim = kS;
  v8f o = {0.f, 0.f, 0.f, 0.f, 0.f, 0.f, 0.f, 0.f};
  for (int kk = 0; kk < klim; kk += 32) {
    Frag pf, vf;
    pf.h[0] = *(const v8h*)&pb[ln][kk + g * 16];
    pf.h[1] = *(const v8h*)&pb[ln][kk + g * 16 + 8];
#pragma unroll
    for (int e = 0; e < 16; ++e) {
      int tok = b * kS + kk + 16 * g + e;
      vf.v[e] = vb[((size_t)tok * kNKV + kvh) * kHD + hd];
    }
    o = __builtin_amdgcn_wmma_f32_16x16x32_f16(
        false, pf.v, false, vf.v, (short)0, o, false, false);
  }
#pragma unroll
  for (int i = 0; i < 8; ++i) {
    int tok = b * kS + q0 + g * 8 + i;
    ao[(size_t)tok * kD + hh * kHD + hd] = (_Float16)o[i];
  }
}

// ---------------- silu(gate) * up ----------------
__global__ void silu_mul_kernel(const _Float16* g, const _Float16* u,
                                _Float16* out, size_t n) {
  size_t i = (size_t)blockIdx.x * blockDim.x + threadIdx.x;
  if (i < n) {
    float gv = (float)g[i];
    float s = gv / (1.f + __expf(-gv));
    out[i] = (_Float16)(s * (float)u[i]);
  }
}

// ---------------- pooled head: normalize, dot-product, pairwise mul -------
__global__ __launch_bounds__(256) void head_kernel(
    const float* __restrict__ h, const int* __restrict__ amask,
    const float* __restrict__ dpw, const float* __restrict__ dpb,
    float* __restrict__ out) {
  __shared__ int lastIdx[kB];
  __shared__ float invn[kB];
  __shared__ float semb[kB][50];
  int tid = threadIdx.x, lane = tid & 31, wv = tid >> 5;
  if (wv < kB) {
    int s = 0;
    for (int j = lane; j < kS; j += 32) s += (amask[wv * kS + j] > 0);
    s = waveReduceSumI(s);
    if (lane == 0) lastIdx[wv] = s - 1;
  }
  __syncthreads();
  if (wv < kB) {
    const float* row = h + (size_t)(wv * kS + lastIdx[wv]) * kD;
    float ss = 0.f;
    for (int j = lane; j < kD; j += 32) { float v = row[j]; ss += v * v; }
    ss = waveReduceSum(ss);
    if (lane == 0) invn[wv] = 1.0f / fmaxf(sqrtf(ss), 1e-12f);
  }
  __syncthreads();
  for (int p = wv; p < kB * 50; p += 8) {
    int bb = p / 50, j = p % 50;
    const float* row = h + (size_t)(bb * kS + lastIdx[bb]) * kD;
    const float* wr = dpw + (size_t)j * kD;
    float dot = 0.f;
    for (int i = lane; i < kD; i += 32) dot += row[i] * wr[i];
    dot = waveReduceSum(dot);
    if (lane == 0) semb[bb][j] = dot * invn[bb] + dpb[j];
  }
  __syncthreads();
  if (tid < 100) {
    int bb = tid / 50, j = tid % 50;
    out[tid] = semb[bb][j] * semb[bb + 2][j];
  }
}

// ---------------- launch ----------------
extern "C" void kernel_launch(void* const* d_in, const int* in_sizes, int n_in,
                              void* d_out, int out_size, void* d_ws,
                              size_t ws_size, hipStream_t stream) {
  const int*   input_ids = (const int*)d_in[0];
  const int*   amask     = (const int*)d_in[1];
  const float* etab      = (const float*)d_in[2];
  const int*   idx_q = (const int*)d_in[3];
  const int*   idx_k = (const int*)d_in[4];
  const int*   idx_v = (const int*)d_in[5];
  const int*   idx_o = (const int*)d_in[6];
  const int*   idx_g = (const int*)d_in[7];
  const int*   idx_u = (const int*)d_in[8];
  const int*   idx_d = (const int*)d_in[9];
  const float* am_q = (const float*)d_in[10];
  const float* am_k = (const float*)d_in[11];
  const float* am_v = (const float*)d_in[12];
  const float* am_o = (const float*)d_in[13];
  const float* am_g = (const float*)d_in[14];
  const float* am_u = (const float*)d_in[15];
  const float* am_d = (const float*)d_in[16];
  const float* rms1 = (const float*)d_in[17];
  const float* rms2 = (const float*)d_in[18];
  const float* dpw  = (const float*)d_in[19];
  const float* dpb  = (const float*)d_in[20];

  char* ws = (char*)d_ws;
  size_t off = 0;
  auto carve = [&](size_t bytes) {
    char* p = ws + off;
    off = (off + bytes + 255) & ~(size_t)255;
    return p;
  };
  float*     h  = (float*)carve((size_t)kTok * kD * 4);       // residual
  _Float16*  r  = (_Float16*)carve((size_t)kTok * kD * 2);    // rmsnorm out
  _Float16*  qb = (_Float16*)carve((size_t)kTok * kD * 2);
  _Float16*  kbuf = (_Float16*)carve((size_t)kTok * kDKV * 2);
  _Float16*  vbuf = (_Float16*)carve((size_t)kTok * kDKV * 2);
  _Float16*  ao = (_Float16*)carve((size_t)kTok * kD * 2);
  _Float16*  gb = (_Float16*)carve((size_t)kTok * kDFF * 2);
  _Float16*  ub = (_Float16*)carve((size_t)kTok * kDFF * 2);

  embed_kernel<<<kTok, 256, 0, stream>>>(input_ids, etab, h);
  rmsnorm_kernel<<<kTok, 256, 0, stream>>>(h, rms1, r);

  // QKV projections (NF4 dequant + WMMA, async LDS staging)
  gemm_nf4_kernel<<<dim3(kD / BN, kTok / BM), 256, 0, stream>>>(
      r, idx_q, am_q, qb, kD, kD, 0);
  gemm_nf4_kernel<<<dim3(kDKV / BN, kTok / BM), 256, 0, stream>>>(
      r, idx_k, am_k, kbuf, kDKV, kD, 0);
  gemm_nf4_kernel<<<dim3(kDKV / BN, kTok / BM), 256, 0, stream>>>(
      r, idx_v, am_v, vbuf, kDKV, kD, 0);

  rope_kernel<<<dim3(kTok, kNH), 64, 0, stream>>>(qb, kNH);
  rope_kernel<<<dim3(kTok, kNKV), 64, 0, stream>>>(kbuf, kNKV);

  attn_kernel<<<dim3(kS / 16, kNH, kB), 256, 0, stream>>>(qb, kbuf, vbuf,
                                                          amask, ao);
  // output projection, accumulate into residual
  gemm_nf4_kernel<<<dim3(kD / BN, kTok / BM), 256, 0, stream>>>(
      ao, idx_o, am_o, h, kD, kD, 1);

  rmsnorm_kernel<<<kTok, 256, 0, stream>>>(h, rms2, r);

  gemm_nf4_kernel<<<dim3(kDFF / BN, kTok / BM), 256, 0, stream>>>(
      r, idx_g, am_g, gb, kDFF, kD, 0);
  gemm_nf4_kernel<<<dim3(kDFF / BN, kTok / BM), 256, 0, stream>>>(
      r, idx_u, am_u, ub, kDFF, kD, 0);

  size_t nff = (size_t)kTok * kDFF;
  silu_mul_kernel<<<(unsigned)((nff + 255) / 256), 256, 0, stream>>>(gb, ub,
                                                                     gb, nff);
  gemm_nf4_kernel<<<dim3(kD / BN, kTok / BM), 256, 0, stream>>>(
      gb, idx_d, am_d, h, kD, kDFF, 1);

  head_kernel<<<1, 256, 0, stream>>>(h, amask, dpw, dpb, (float*)d_out);
}
